// KGramMLPSeqModel_71305047048767
// MI455X (gfx1250) — matrix operand order repaired and checked by reading, not compiled, not run on hardware
//
#include <hip/hip_runtime.h>
#include <hip/hip_bf16.h>

// ---------------- problem constants ----------------
#define VOCAB 32000
#define KGRAM 3
#define EMBED 1024
#define SEQ   2048
#define BATCH 2
#define ROWS  (SEQ * BATCH)          // 4096 rows of H / logits
#define MTILES (ROWS / 16)           // 256 M tiles
#define NTILES_N (VOCAB / 16)        // 2000 N tiles

#define NT  8                        // N-tiles (of 16) per wave -> 128 cols
#define WPB 8                        // waves per block, each a distinct M tile
#define NGROUPS (NTILES_N / NT)      // 250 N groups
#define MGROUPS (MTILES / WPB)       // 32 M groups

typedef __attribute__((ext_vector_type(16))) _Float16 v16h;
typedef __attribute__((ext_vector_type(8)))  _Float16 v8h;
typedef __attribute__((ext_vector_type(4)))  _Float16 v4h;
typedef __attribute__((ext_vector_type(8)))  float    v8f;
typedef __attribute__((ext_vector_type(4)))  float    v4f;

// ---------------------------------------------------------------------------
// Kernel 0: W2 (EMBED x VOCAB, f32, row-major) -> W2T (VOCAB x EMBED, f16)
// LDS-tiled transpose; both read and write coalesced.
// grid: (VOCAB/32, EMBED/32), block: (32, 8)
// ---------------------------------------------------------------------------
__global__ __launch_bounds__(256)
void w2_transpose_f16(const float* __restrict__ W2, _Float16* __restrict__ W2T) {
    __shared__ float tile[32][33];
    const int bx = blockIdx.x;   // n / 32
    const int by = blockIdx.y;   // e / 32
    const int tx = threadIdx.x;  // 0..31
    const int ty = threadIdx.y;  // 0..7

#pragma unroll
    for (int i = 0; i < 4; ++i) {
        int e = by * 32 + ty + 8 * i;
        int n = bx * 32 + tx;
        tile[ty + 8 * i][tx] = W2[(size_t)e * VOCAB + n];
    }
    __syncthreads();
#pragma unroll
    for (int i = 0; i < 4; ++i) {
        int n = bx * 32 + ty + 8 * i;
        int e = by * 32 + tx;
        W2T[(size_t)n * EMBED + e] = (_Float16)tile[tx][ty + 8 * i];
    }
}

// ---------------------------------------------------------------------------
// Kernel 1: H[r][e] = silu( sum_j W1t[j*VOCAB + tok_j(r)][e] + b1[e] ), f16
// ---------------------------------------------------------------------------
__global__ __launch_bounds__(256)
void build_h(const int* __restrict__ tokens, const float* __restrict__ W1t,
             const float* __restrict__ b1, _Float16* __restrict__ H) {
    const int r = blockIdx.x;          // 0..4095  (r = t*BATCH + b)
    const int e = threadIdx.x * 4;     // 0..1020
    const int t = r >> 1;
    const int b = r & 1;

    v4f acc = *(const v4f*)(b1 + e);
#pragma unroll
    for (int j = 0; j < KGRAM; ++j) {
        int tp = t + j - (KGRAM - 1);
        int tok = (tp >= 0) ? tokens[tp * BATCH + b] : 0;
        size_t row = (size_t)(j * VOCAB + tok);
        v4f w = *(const v4f*)(W1t + row * EMBED + e);
        acc += w;
    }
    v4h h;
#pragma unroll
    for (int i = 0; i < 4; ++i) {
        float x = acc[i];
        float s = x / (1.0f + __expf(-x));   // SiLU
        h[i] = (_Float16)s;
    }
    *(v4h*)(H + (size_t)r * EMBED + e) = h;
}

// ---------------------------------------------------------------------------
// Kernel 2: logits = H x W2T^T + b2 via v_wmma_f32_16x16x32_f16.
//
// Block = 8 waves. All waves share the SAME 128 output columns (same B
// fragments -> WGP$/L0 hits after the first wave) and cover 8 distinct M
// tiles. M-group is the fast block index so concurrent blocks also share B
// in time. Per k-slice: 2 A loads + 16 B loads issued as one clause before
// the 8-WMMA group -> descending partial s_wait_loadcnt instead of 0-drains.
//
// Fragment layouts used here (A/B: contiguous K order, same for both):
//  A (16x32 f16): lanes 0-15 -> M=lane, K = {k..k+15}
//                 lanes 16-31 -> M=lane-16, K = {k+16..k+31}
//  B (32x16 f16): lane -> column N = lane%16, K-half = 16*(lane/16)
//  C/D (16x16 f32): element v of lane L -> M = v + 8*(L/16), N = L%16
// ---------------------------------------------------------------------------
__global__ __launch_bounds__(256)
void gemm_wmma(const _Float16* __restrict__ H, const _Float16* __restrict__ W2T,
               const float* __restrict__ b2, float* __restrict__ out) {
    const int w    = threadIdx.x >> 5;   // wave in block: 0..7
    const int lane = threadIdx.x & 31;
    const int lm   = lane & 15;          // M row (A) / N col (B,C)
    const int lh   = lane >> 4;          // half selector

    const int mg = blockIdx.x % MGROUPS;         // fast: M group (B stays hot)
    const int ng = blockIdx.x / MGROUPS;         // slow: N group
    const int tileM = mg * WPB + w;              // 0..255
    const int nbase = ng * (NT * 16);            // 0..31872

    // A: row = tileM*16 + lm, per-lane K sub-offset 16*lh
    const _Float16* arow = H + (size_t)(tileM * 16 + lm) * EMBED + lh * 16;
    // B: base column pointer; tile j at constant offset j*16*EMBED (imm offsets)
    const _Float16* bcol = W2T + (size_t)(nbase + lm) * EMBED + lh * 16;

    v8f acc[NT];
#pragma unroll
    for (int j = 0; j < NT; ++j) acc[j] = (v8f)0.0f;

    union Au { v16h v; v8h h[2]; };

#pragma unroll 2
    for (int k = 0; k < EMBED; k += 32) {
        // ---- load full k-slice fragments first (single load clause) ----
        Au a;
        a.h[0] = *(const v8h*)(arow + k);        // K = k+16*lh .. +8
        a.h[1] = *(const v8h*)(arow + k + 8);    // K = k+8+16*lh .. +8
        v16h bf[NT];
#pragma unroll
        for (int j = 0; j < NT; ++j)
            bf[j] = *(const v16h*)(bcol + (size_t)j * 16 * EMBED + k);

        // ---- 8 WMMAs consuming the slice ----
#pragma unroll
        for (int j = 0; j < NT; ++j)
            acc[j] = __builtin_amdgcn_wmma_f32_16x16x32_f16(
                false, a.v, false, bf[j], (short)0, acc[j], false, false);
    }

    // bias + store: lanes 0-15 form contiguous 64B row segments per VGPR
#pragma unroll
    for (int j = 0; j < NT; ++j) {
        int n = nbase + j * 16 + lm;
        float bias = b2[n];
#pragma unroll
        for (int v = 0; v < 8; ++v) {
            int m = tileM * 16 + v + 8 * lh;
            out[(size_t)m * VOCAB + n] = acc[j][v] + bias;
        }
    }
}

// ---------------------------------------------------------------------------
extern "C" void kernel_launch(void* const* d_in, const int* in_sizes, int n_in,
                              void* d_out, int out_size, void* d_ws, size_t ws_size,
                              hipStream_t stream) {
    const int*   tokens = (const int*)  d_in[0];  // (SEQ, BATCH)
    const float* W1t    = (const float*)d_in[1];  // (K*VOCAB, EMBED)
    const float* b1     = (const float*)d_in[2];  // (EMBED,)
    const float* W2     = (const float*)d_in[3];  // (EMBED, VOCAB)
    const float* b2     = (const float*)d_in[4];  // (VOCAB,)
    float* out = (float*)d_out;                   // (SEQ, BATCH, VOCAB)

    _Float16* W2T = (_Float16*)d_ws;                                      // 65,536,000 B
    _Float16* H   = (_Float16*)((char*)d_ws + (size_t)VOCAB * EMBED * 2); // 8,388,608 B

    // Kernel 0: transpose + f32->f16 convert of W2
    dim3 tgrid(VOCAB / 32, EMBED / 32);
    dim3 tblk(32, 8);
    w2_transpose_f16<<<tgrid, tblk, 0, stream>>>(W2, W2T);

    // Kernel 1: gather + bias + SiLU -> H (f16)
    build_h<<<ROWS, 256, 0, stream>>>(tokens, W1t, b1, H);

    // Kernel 2: WMMA GEMM + bias -> logits
    gemm_wmma<<<MGROUPS * NGROUPS, 256, 0, stream>>>(H, W2T, b2, out);
}
